// SigGateGPSLayer_83391085019438
// MI455X (gfx1250) — matrix operand order repaired and checked by reading, not compiled, not run on hardware
//
#include <hip/hip_runtime.h>
#include <math.h>

// ---------------- problem constants ----------------
#define N_   32768
#define E_   262144
#define D_   256
#define H_   8
#define G_   64
#define NPG_ 512
#define FFN_ 512
#define DH_  32

// ---------------- types ----------------
typedef __bf16 bf16_t;
typedef __attribute__((ext_vector_type(16))) __bf16 v16bf;
typedef __attribute__((ext_vector_type(8)))  __bf16 v8bf;
typedef __attribute__((ext_vector_type(8)))  float  v8f;

union ABfrag { v16bf v; v8bf h[2]; };

__device__ __forceinline__ bf16_t f2bf(float f) {
    union { float f; unsigned u; } c; c.f = f;
    unsigned r = c.u + 0x7FFFu + ((c.u >> 16) & 1u);           // RNE
    union { unsigned short s; bf16_t b; } o; o.s = (unsigned short)(r >> 16);
    return o.b;
}
__device__ __forceinline__ float sigm(float x) { return 1.0f / (1.0f + __expf(-x)); }

// ---------------- generic f32 -> bf16 convert ----------------
__global__ __launch_bounds__(256) void k_f2bf(const float* __restrict__ x,
                                              bf16_t* __restrict__ y, size_t n) {
    size_t i = (size_t)blockIdx.x * 256 + threadIdx.x;
    if (i < n) y[i] = f2bf(x[i]);
}

// ---------------- GEMM: C[M,Nn] = A[M,K](bf16) @ W[Nn,K]^T(bf16) + bias ----------------
// one wave -> one 16x16 tile; 8 waves/block -> 32 rows x 64 cols
__global__ __launch_bounds__(256)
void k_gemm_tn(const bf16_t* __restrict__ A, const bf16_t* __restrict__ W,
               const float* __restrict__ bias, float* __restrict__ C,
               int M, int Nn, int K) {
    const int lane  = threadIdx.x & 31;
    const int wave  = threadIdx.x >> 5;
    const int tileN = blockIdx.x * 64 + (wave & 3) * 16;
    const int tileM = blockIdx.y * 32 + (wave >> 2) * 16;
    if (tileM >= M || tileN >= Nn) return;          // wave-uniform
    const int r     = lane & 15;
    const int khalf = lane >> 4;
    const bf16_t* arow = A + (size_t)(tileM + r) * K + khalf * 8;   // A: k 0-7 / +16..23
    const bf16_t* wrow = W + (size_t)(tileN + r) * K + khalf * 8;   // B: k 0-7 / +16..23
    v8f acc = {};
    for (int k0 = 0; k0 < K; k0 += 32) {
        ABfrag a, b;
        a.h[0] = *(const v8bf*)(arow + k0);
        a.h[1] = *(const v8bf*)(arow + k0 + 16);
        b.h[0] = *(const v8bf*)(wrow + k0);
        b.h[1] = *(const v8bf*)(wrow + k0 + 16);
        acc = __builtin_amdgcn_wmma_f32_16x16x32_bf16(false, a.v, false, b.v,
                                                      (short)0, acc, false, false);
    }
    const int cn = tileN + r;
    const float bv = bias ? bias[cn] : 0.0f;
    float* crow = C + (size_t)(tileM + khalf * 8) * Nn + cn;
    #pragma unroll
    for (int v = 0; v < 8; ++v) crow[(size_t)v * Nn] = acc[v] + bv;
}

// ---------------- attention scores: S[z,q,k] = (Q . K)/sqrt(DH), z = g*H+h ----------------
__global__ __launch_bounds__(256)
void k_attn_scores(const bf16_t* __restrict__ Q, const bf16_t* __restrict__ Kb,
                   float* __restrict__ S, float scale) {
    const int z = blockIdx.z, g = z >> 3, h = z & 7;
    const int lane = threadIdx.x & 31, wave = threadIdx.x >> 5;
    const int tileN = blockIdx.x * 64 + (wave & 3) * 16;   // key idx
    const int tileM = blockIdx.y * 32 + (wave >> 2) * 16;  // query idx
    const int r = lane & 15, khalf = lane >> 4;
    const bf16_t* q  = Q  + ((size_t)(g * NPG_ + tileM + r)) * D_ + h * DH_ + khalf * 8;
    const bf16_t* kk = Kb + ((size_t)(g * NPG_ + tileN + r)) * D_ + h * DH_ + khalf * 8;
    ABfrag a, b;
    a.h[0] = *(const v8bf*)q;        a.h[1] = *(const v8bf*)(q + 16);
    b.h[0] = *(const v8bf*)kk;       b.h[1] = *(const v8bf*)(kk + 16);
    v8f acc = {};
    acc = __builtin_amdgcn_wmma_f32_16x16x32_bf16(false, a.v, false, b.v,
                                                  (short)0, acc, false, false);
    float* srow = S + ((size_t)z * NPG_ + tileM + khalf * 8) * NPG_ + tileN + r;
    #pragma unroll
    for (int v = 0; v < 8; ++v) srow[(size_t)v * NPG_] = acc[v] * scale;
}

// ---------------- softmax over 512-wide rows, write bf16 ----------------
__global__ __launch_bounds__(256)
void k_softmax_bf16(const float* __restrict__ S, bf16_t* __restrict__ P) {
    __shared__ float red[256];
    const size_t row = blockIdx.x;
    const float* s = S + row * (size_t)NPG_;
    bf16_t* p = P + row * (size_t)NPG_;
    const int t = threadIdx.x;
    float a = s[t], b = s[t + 256];
    red[t] = fmaxf(a, b); __syncthreads();
    for (int o = 128; o > 0; o >>= 1) { if (t < o) red[t] = fmaxf(red[t], red[t + o]); __syncthreads(); }
    float m = red[0]; __syncthreads();
    float ea = __expf(a - m), eb = __expf(b - m);
    red[t] = ea + eb; __syncthreads();
    for (int o = 128; o > 0; o >>= 1) { if (t < o) red[t] += red[t + o]; __syncthreads(); }
    float inv = 1.0f / red[0];
    p[t] = f2bf(ea * inv); p[t + 256] = f2bf(eb * inv);
}

// ---------------- V transpose: vt[z][dh][key] = v[g*NPG+key][h*DH+dh] (bf16) ----------------
__global__ __launch_bounds__(256)
void k_vt(const float* __restrict__ vlin, bf16_t* __restrict__ vt) {
    size_t i = (size_t)blockIdx.x * 256 + threadIdx.x;   // output-major -> coalesced writes
    int    p  = (int)(i % NPG_);
    size_t zd = i / NPG_;
    int    dh = (int)(zd % DH_);
    int    z  = (int)(zd / DH_);
    int    g  = z >> 3, h = z & 7;
    vt[i] = f2bf(vlin[((size_t)(g * NPG_ + p)) * D_ + h * DH_ + dh]);
}

// ---------------- O[z,q,dh] = P[z,q,:] @ VT[z,dh,:]^T  (pure TN, contiguous loads) -------
__global__ __launch_bounds__(256)
void k_attn_av(const bf16_t* __restrict__ P, const bf16_t* __restrict__ VT,
               float* __restrict__ O) {
    const int z = blockIdx.z, g = z >> 3, h = z & 7;
    const int lane = threadIdx.x & 31, wave = threadIdx.x >> 5;
    const int tileM = blockIdx.y * 64 + (wave >> 1) * 16;  // 4 row-tiles
    const int tileN = (wave & 1) * 16;                     // 2 col-tiles (DH=32)
    const int r = lane & 15, khalf = lane >> 4;
    const bf16_t* arow = P  + ((size_t)z * NPG_ + tileM + r) * NPG_ + khalf * 8;
    const bf16_t* vrow = VT + ((size_t)z * DH_ + tileN + r) * NPG_ + khalf * 8;
    v8f acc = {};
    for (int k0 = 0; k0 < NPG_; k0 += 32) {
        ABfrag a, b;
        a.h[0] = *(const v8bf*)(arow + k0);
        a.h[1] = *(const v8bf*)(arow + k0 + 16);
        b.h[0] = *(const v8bf*)(vrow + k0);
        b.h[1] = *(const v8bf*)(vrow + k0 + 16);
        acc = __builtin_amdgcn_wmma_f32_16x16x32_bf16(false, a.v, false, b.v,
                                                      (short)0, acc, false, false);
    }
    float* orow = O + ((size_t)(g * NPG_ + tileM + khalf * 8)) * D_ + h * DH_ + tileN + r;
    #pragma unroll
    for (int v = 0; v < 8; ++v) orow[(size_t)v * D_] = acc[v];
}

// ---------------- edge gate + scatter (segment sums via f32 atomics) ----------------
__global__ __launch_bounds__(256)
void k_edge_gate(const float* __restrict__ xs, const float* __restrict__ xd,
                 const float* __restrict__ xm, const float* __restrict__ eW,
                 const int* __restrict__ src, const int* __restrict__ dst,
                 float* __restrict__ gate, bf16_t* __restrict__ gate_bf,
                 float* __restrict__ gsum, float* __restrict__ agg) {
    const int e = blockIdx.x, d = threadIdx.x;
    const int s = src[e], t = dst[e];
    float gl = xs[(size_t)s * D_ + d] + xd[(size_t)t * D_ + d] + eW[(size_t)e * D_ + d];
    float gv = sigm(gl);
    gate[(size_t)e * D_ + d]    = gv;
    gate_bf[(size_t)e * D_ + d] = f2bf(gv);
    atomicAdd(&gsum[(size_t)t * D_ + d], gv);
    atomicAdd(&agg[(size_t)t * D_ + d], xm[(size_t)s * D_ + d] * gv);
}

// ---------------- pre = xr + agg/(gsum+1e-6) ----------------
__global__ __launch_bounds__(256)
void k_pre(const float* __restrict__ xr, const float* __restrict__ agg,
           const float* __restrict__ gsum, float* __restrict__ pre) {
    size_t i = (size_t)blockIdx.x * 256 + threadIdx.x;
    pre[i] = xr[i] + agg[i] / (gsum[i] + 1e-6f);
}

// ---------------- BN column stats, coalesced: block owns 256 rows; thread d owns feature d
__global__ __launch_bounds__(256)
void k_colpart(const float* __restrict__ X, float* __restrict__ sums) {
    const int d  = threadIdx.x;
    const size_t r0 = (size_t)blockIdx.x * 256;
    float s = 0.f, sq = 0.f;
    for (int r = 0; r < 256; ++r) {
        float v = X[(r0 + r) * D_ + d];          // row-major, fully coalesced
        s += v; sq += v * v;
    }
    atomicAdd(&sums[d], s);
    atomicAdd(&sums[D_ + d], sq);
}

__global__ __launch_bounds__(256)
void k_colfin(const float* __restrict__ sums, float* __restrict__ stats, float invR) {
    const int d = threadIdx.x;
    float m = sums[d] * invR;
    float v = sums[D_ + d] * invR - m * m;
    stats[d] = m;
    stats[D_ + d] = 1.0f / sqrtf(v + 1e-5f);
}

// ---------------- node: relu(BN(pre)) + x residual, then LayerNorm -> x1 (f32 + bf16) ----------------
__global__ __launch_bounds__(256)
void k_node_bn_ln(const float* __restrict__ x, const float* __restrict__ pre,
                  const float* __restrict__ stats,
                  const float* __restrict__ bng, const float* __restrict__ bnb,
                  const float* __restrict__ lng, const float* __restrict__ lnb,
                  float* __restrict__ x1, bf16_t* __restrict__ x1bf) {
    const int n = blockIdx.x, d = threadIdx.x;
    float p  = pre[(size_t)n * D_ + d];
    float bn = fmaxf(bng[d] * (p - stats[d]) * stats[D_ + d] + bnb[d], 0.0f);
    float t  = x[(size_t)n * D_ + d] + bn;
    __shared__ float rs[256], rq[256];
    rs[d] = t; rq[d] = t * t; __syncthreads();
    for (int o = 128; o > 0; o >>= 1) {
        if (d < o) { rs[d] += rs[d + o]; rq[d] += rq[d + o]; }
        __syncthreads();
    }
    float m = rs[0] / (float)D_;
    float rstd = 1.0f / sqrtf(rq[0] / (float)D_ - m * m + 1e-5f);
    float y = lng[d] * (t - m) * rstd + lnb[d];
    x1[(size_t)n * D_ + d] = y;
    x1bf[(size_t)n * D_ + d] = f2bf(y);
}

// ---------------- generic residual + LayerNorm (optionally also bf16 copy) ----------------
__global__ __launch_bounds__(256)
void k_res_ln(const float* __restrict__ a, const float* __restrict__ b,
              const float* __restrict__ lng, const float* __restrict__ lnb,
              float* __restrict__ out, bf16_t* __restrict__ outbf) {
    const int n = blockIdx.x, d = threadIdx.x;
    float t = a[(size_t)n * D_ + d] + b[(size_t)n * D_ + d];
    __shared__ float rs[256], rq[256];
    rs[d] = t; rq[d] = t * t; __syncthreads();
    for (int o = 128; o > 0; o >>= 1) {
        if (d < o) { rs[d] += rs[d + o]; rq[d] += rq[d + o]; }
        __syncthreads();
    }
    float m = rs[0] / (float)D_;
    float rstd = 1.0f / sqrtf(rq[0] / (float)D_ - m * m + 1e-5f);
    float y = lng[d] * (t - m) * rstd + lnb[d];
    out[(size_t)n * D_ + d] = y;
    if (outbf) outbf[(size_t)n * D_ + d] = f2bf(y);
}

// ---------------- edge output: relu(BN(elin)) ----------------
__global__ __launch_bounds__(256)
void k_edge_bn(const float* __restrict__ elin, const float* __restrict__ stats,
               const float* __restrict__ g, const float* __restrict__ b,
               float* __restrict__ out) {
    const int e = blockIdx.x, d = threadIdx.x;
    float v = elin[(size_t)e * D_ + d];
    out[(size_t)e * D_ + d] = fmaxf(g[d] * (v - stats[d]) * stats[D_ + d] + b[d], 0.0f);
}

// ---------------- og = bf16(o * sigmoid(glin)) ----------------
__global__ __launch_bounds__(256)
void k_og(const float* __restrict__ o, const float* __restrict__ glin,
          bf16_t* __restrict__ ogbf) {
    size_t i = (size_t)blockIdx.x * 256 + threadIdx.x;
    ogbf[i] = f2bf(o[i] * sigm(glin[i]));
}

// ---------------- exact GELU -> bf16 ----------------
__global__ __launch_bounds__(256)
void k_gelu_bf(const float* __restrict__ x, bf16_t* __restrict__ y) {
    size_t i = (size_t)blockIdx.x * 256 + threadIdx.x;
    float v = x[i];
    y[i] = f2bf(0.5f * v * (1.0f + erff(v * 0.70710678118654752f)));
}

// =====================================================================
extern "C" void kernel_launch(void* const* d_in, const int* in_sizes, int n_in,
                              void* d_out, int out_size, void* d_ws, size_t ws_size,
                              hipStream_t stream) {
    (void)in_sizes; (void)n_in; (void)out_size; (void)ws_size;
    const float* x   = (const float*)d_in[0];
    const float* ea  = (const float*)d_in[1];
    const float *Ws = (const float*)d_in[2],  *bs = (const float*)d_in[3];
    const float *Wd = (const float*)d_in[4],  *bd = (const float*)d_in[5];
    const float *We = (const float*)d_in[6],  *be = (const float*)d_in[7];
    const float *Wm = (const float*)d_in[8],  *bm = (const float*)d_in[9];
    const float *Wr = (const float*)d_in[10], *br = (const float*)d_in[11];
    const float *Weo= (const float*)d_in[12], *beo= (const float*)d_in[13];
    const float *bnng=(const float*)d_in[14], *bnnb=(const float*)d_in[15];
    const float *bneg=(const float*)d_in[16], *bneb=(const float*)d_in[17];
    const float *Wq = (const float*)d_in[18], *bq = (const float*)d_in[19];
    const float *Wk = (const float*)d_in[20], *bk = (const float*)d_in[21];
    const float *Wv = (const float*)d_in[22], *bv = (const float*)d_in[23];
    const float *Wg = (const float*)d_in[24], *bg = (const float*)d_in[25];
    const float *Wo = (const float*)d_in[26], *bo = (const float*)d_in[27];
    const float *ln1g=(const float*)d_in[28], *ln1b=(const float*)d_in[29];
    const float *ln2g=(const float*)d_in[30], *ln2b=(const float*)d_in[31];
    const float *ln3g=(const float*)d_in[32], *ln3b=(const float*)d_in[33];
    const float *W1 = (const float*)d_in[34], *b1 = (const float*)d_in[35];
    const float *W2 = (const float*)d_in[36], *b2 = (const float*)d_in[37];
    const int*   ei = (const int*)d_in[38];
    const int *src = ei, *dst = ei + E_;

    // ---- workspace bump allocator ----
    char* ws = (char*)d_ws;
    size_t off = 0;
    auto alloc = [&](size_t bytes) -> void* {
        off = (off + 255) & ~(size_t)255;
        void* p = ws + off; off += bytes; return p;
    };
    const size_t ND4 = (size_t)N_ * D_ * 4, ND2 = (size_t)N_ * D_ * 2;
    const size_t ED4 = (size_t)E_ * D_ * 4, ED2 = (size_t)E_ * D_ * 2;
    const size_t S4  = (size_t)G_ * H_ * NPG_ * NPG_ * 4;

    // bf16 weights
    bf16_t *wsb=(bf16_t*)alloc(D_*D_*2), *wdb=(bf16_t*)alloc(D_*D_*2), *web=(bf16_t*)alloc(D_*D_*2),
           *wmb=(bf16_t*)alloc(D_*D_*2), *wrb=(bf16_t*)alloc(D_*D_*2), *weob=(bf16_t*)alloc(D_*D_*2),
           *wqb=(bf16_t*)alloc(D_*D_*2), *wkb=(bf16_t*)alloc(D_*D_*2), *wvb=(bf16_t*)alloc(D_*D_*2),
           *wgb=(bf16_t*)alloc(D_*D_*2), *wob=(bf16_t*)alloc(D_*D_*2),
           *w1b=(bf16_t*)alloc((size_t)FFN_*D_*2), *w2b=(bf16_t*)alloc((size_t)D_*FFN_*2);
    float* statsN = (float*)alloc(2 * D_ * 4);
    float* statsE = (float*)alloc(2 * D_ * 4);
    float* sumsN  = (float*)alloc(2 * D_ * 4);     // adjacent to sumsE -> single memset
    float* sumsE  = (float*)alloc(2 * D_ * 4);
    bf16_t* xbf = (bf16_t*)alloc(ND2);
    float *xs=(float*)alloc(ND4), *xd=(float*)alloc(ND4), *xm=(float*)alloc(ND4), *xr=(float*)alloc(ND4);
    float *gsum=(float*)alloc(ND4), *agg=(float*)alloc(ND4);   // adjacent -> single memset
    float *pre=(float*)alloc(ND4);
    float *x1=(float*)alloc(ND4); bf16_t* x1bf=(bf16_t*)alloc(ND2);
    float *qlin=(float*)alloc(ND4), *klin=(float*)alloc(ND4), *vlin=(float*)alloc(ND4), *glin=(float*)alloc(ND4);
    bf16_t *qbf=(bf16_t*)alloc(ND2), *kbf=(bf16_t*)alloc(ND2), *vt=(bf16_t*)alloc(ND2);
    float *o=(float*)alloc(ND4); bf16_t* ogbf=(bf16_t*)alloc(ND2);
    float *attn_out=(float*)alloc(ND4);
    float *x2=(float*)alloc(ND4); bf16_t* x2bf=(bf16_t*)alloc(ND2);
    float *ffn1=(float*)alloc((size_t)N_*FFN_*4); bf16_t* ffn1bf=(bf16_t*)alloc((size_t)N_*FFN_*2);
    float *ffn2=(float*)alloc(ND4);
    // edge slab (reused as scores + attn weights after edge output is written)
    char* slab = (char*)alloc(ED2 + 2 * ED4 + ED2);            // 768 MB
    bf16_t* ebf    = (bf16_t*)slab;
    float*  eW     = (float*)(slab + ED2);
    float*  gate   = (float*)(slab + ED2 + ED4);
    bf16_t* gatebf = (bf16_t*)(slab + ED2 + 2 * ED4);
    float*  elin   = eW;                                        // eW consumed before elin written
    float*  scores = (float*)slab;                              // after edge path done
    bf16_t* attnbf = (bf16_t*)(slab + S4);

    float* out_x3   = (float*)d_out;
    float* out_edge = (float*)d_out + (size_t)N_ * D_;

    auto gemm = [&](const bf16_t* A, const bf16_t* Wt, const float* bias, float* C,
                    int M, int Nn, int K) {
        dim3 g(Nn / 64, M / 32, 1);
        k_gemm_tn<<<g, dim3(256), 0, stream>>>(A, Wt, bias, C, M, Nn, K);
    };
    auto conv = [&](const float* a, bf16_t* b, size_t n) {
        k_f2bf<<<dim3((unsigned)((n + 255) / 256)), dim3(256), 0, stream>>>(a, b, n);
    };

    // ---- convert weights & activations to bf16 ----
    conv(Ws, wsb, D_*D_);  conv(Wd, wdb, D_*D_);  conv(We, web, D_*D_);
    conv(Wm, wmb, D_*D_);  conv(Wr, wrb, D_*D_);  conv(Weo, weob, D_*D_);
    conv(Wq, wqb, D_*D_);  conv(Wk, wkb, D_*D_);  conv(Wv, wvb, D_*D_);
    conv(Wg, wgb, D_*D_);  conv(Wo, wob, D_*D_);
    conv(W1, w1b, (size_t)FFN_*D_); conv(W2, w2b, (size_t)D_*FFN_);
    conv(x, xbf, (size_t)N_*D_);
    conv(ea, ebf, (size_t)E_*D_);

    // ---- zero accumulators (sumsN+sumsE contiguous, gsum+agg contiguous) ----
    hipMemsetAsync(sumsN, 0, 4 * D_ * 4, stream);
    hipMemsetAsync(gsum, 0, 2 * ND4, stream);

    // ---- MPNN node-side linears ----
    gemm(xbf, wsb, bs, xs, N_, D_, D_);
    gemm(xbf, wdb, bd, xd, N_, D_, D_);
    gemm(xbf, wmb, bm, xm, N_, D_, D_);
    gemm(xbf, wrb, br, xr, N_, D_, D_);
    // edge linear
    gemm(ebf, web, be, eW, E_, D_, D_);

    // ---- gate + segment sums ----
    k_edge_gate<<<dim3(E_), dim3(256), 0, stream>>>(xs, xd, xm, eW, src, dst,
                                                    gate, gatebf, gsum, agg);

    // ---- edge output branch: relu(BN(gate @ Weo^T + beo)) ----
    gemm(gatebf, weob, beo, elin, E_, D_, D_);
    k_colpart<<<dim3(E_ / 256), dim3(256), 0, stream>>>(elin, sumsE);
    k_colfin<<<dim3(1), dim3(256), 0, stream>>>(sumsE, statsE, 1.0f / (float)E_);
    k_edge_bn<<<dim3(E_), dim3(256), 0, stream>>>(elin, statsE, bneg, bneb, out_edge);

    // ---- node update: pre -> BN -> relu -> +x -> LN1 -> x1 ----
    k_pre<<<dim3(N_ * D_ / 256), dim3(256), 0, stream>>>(xr, agg, gsum, pre);
    k_colpart<<<dim3(N_ / 256), dim3(256), 0, stream>>>(pre, sumsN);
    k_colfin<<<dim3(1), dim3(256), 0, stream>>>(sumsN, statsN, 1.0f / (float)N_);
    k_node_bn_ln<<<dim3(N_), dim3(256), 0, stream>>>(x, pre, statsN, bnng, bnnb,
                                                     ln1g, ln1b, x1, x1bf);

    // ---- attention: q,k,v,gate linears ----
    gemm(x1bf, wqb, bq, qlin, N_, D_, D_);
    gemm(x1bf, wkb, bk, klin, N_, D_, D_);
    gemm(x1bf, wvb, bv, vlin, N_, D_, D_);
    gemm(x1bf, wgb, bg, glin, N_, D_, D_);
    conv(qlin, qbf, (size_t)N_*D_);
    conv(klin, kbf, (size_t)N_*D_);
    k_vt<<<dim3(N_ * D_ / 256), dim3(256), 0, stream>>>(vlin, vt);

    // ---- scores, softmax, attn @ V (edge slab now reused) ----
    const float scale = 1.0f / sqrtf((float)DH_);
    k_attn_scores<<<dim3(NPG_ / 64, NPG_ / 32, G_ * H_), dim3(256), 0, stream>>>(qbf, kbf, scores, scale);
    k_softmax_bf16<<<dim3(G_ * H_ * NPG_), dim3(256), 0, stream>>>(scores, attnbf);
    k_attn_av<<<dim3(1, NPG_ / 64, G_ * H_), dim3(256), 0, stream>>>(attnbf, vt, o);

    // ---- gate, Wo projection, residual + LN2 ----
    k_og<<<dim3(N_ * D_ / 256), dim3(256), 0, stream>>>(o, glin, ogbf);
    gemm(ogbf, wob, bo, attn_out, N_, D_, D_);
    k_res_ln<<<dim3(N_), dim3(256), 0, stream>>>(x1, attn_out, ln2g, ln2b, x2, x2bf);

    // ---- FFN + LN3 -> output ----
    gemm(x2bf, w1b, b1, ffn1, N_, FFN_, D_);
    k_gelu_bf<<<dim3(N_ * FFN_ / 256), dim3(256), 0, stream>>>(ffn1, ffn1bf);
    gemm(ffn1bf, w2b, b2, ffn2, N_, D_, FFN_);
    k_res_ln<<<dim3(N_), dim3(256), 0, stream>>>(x2, ffn2, ln3g, ln3b, out_x3, (bf16_t*)nullptr);
}
